// FeatureRoutingAttention_44985487458310
// MI455X (gfx1250) — matrix operand order repaired and checked by reading, not compiled, not run on hardware
//
#include <hip/hip_runtime.h>
#include <stdint.h>

typedef __attribute__((ext_vector_type(16))) _Float16 v16h;
typedef __attribute__((ext_vector_type(8)))  float    v8f;
typedef __attribute__((ext_vector_type(2)))  float    v2f;
typedef __attribute__((ext_vector_type(4)))  unsigned int u32x4;
typedef __attribute__((ext_vector_type(8)))  unsigned int u32x8;

#define B_  8
#define S_  4096
#define D_  1024
#define E_  8
#define H_  16
#define HD_ 64

union FragU { uint4 q[2]; v16h v; };

__device__ __forceinline__ v8f zero8f() { v8f v; for (int i = 0; i < 8; ++i) v[i] = 0.0f; return v; }
__device__ __forceinline__ v16h zero16h() { v16h v; for (int i = 0; i < 16; ++i) v[i] = (_Float16)0.0f; return v; }

__device__ __forceinline__ v8f wmma_f16(v16h a, v16h b, v8f c) {
  return __builtin_amdgcn_wmma_f32_16x16x32_f16(false, a, false, b, (short)0, c, false, false);
}
__device__ __forceinline__ v8f wmma_f32(v2f a, v2f b, v8f c) {
  return __builtin_amdgcn_wmma_f32_16x16x4_f32(false, a, false, b, (short)0, c, false, false);
}

// Low 32 bits of a flat pointer into LDS == LDS byte offset; the ptrtoint also
// captures the shared object so the asm "memory" clobber is honored (this is
// what keeps the compiler from folding the later ds_load fragments to zero).
__device__ __forceinline__ uint32_t lds_off(const void* p) {
  return (uint32_t)(uintptr_t)p;
}

// CDNA5 async global->LDS copy (16B per lane), tracked with ASYNCcnt.
__device__ __forceinline__ void async_cp16(const void* lds_ptr, const void* gptr) {
  asm volatile("global_load_async_to_lds_b128 %0, %1, off"
               :: "v"(lds_off(lds_ptr)), "v"((unsigned long long)(uintptr_t)gptr)
               : "memory");
}
__device__ __forceinline__ void wait_async0() {
  asm volatile("s_wait_asynccnt 0" ::: "memory");
}

// CDNA5 Tensor Data Mover: 2D tile load, D# groups 0/1 in SGPRs.
__device__ __forceinline__ void tdm_load_2d(const void* lds_ptr, const void* gptr,
                                            uint32_t tensor_d0, uint32_t tensor_d1,
                                            uint32_t tile_d0, uint32_t tile_d1,
                                            uint32_t stride_d0) {
  uint64_t ga = (uint64_t)(uintptr_t)gptr;
  u32x4 g0;
  g0[0] = 1u;                                             // count=1, user mode
  g0[1] = lds_off(lds_ptr);                               // lds_addr
  g0[2] = (uint32_t)ga;                                   // global_addr[31:0]
  g0[3] = (uint32_t)((ga >> 32) & 0x01FFFFFFu) | (2u << 30); // addr[56:32] | type=2
  u32x8 g1;
  g1[0] = (1u << 16);                                     // data_size=1 -> 2 bytes
  g1[1] = (tensor_d0 & 0xFFFFu) << 16;                    // tensor_dim0[15:0]
  g1[2] = ((tensor_d0 >> 16) & 0xFFFFu) | ((tensor_d1 & 0xFFFFu) << 16);
  g1[3] = ((tensor_d1 >> 16) & 0xFFFFu) | ((tile_d0 & 0xFFFFu) << 16);
  g1[4] = (tile_d1 & 0xFFFFu);                            // tile_dim1 | tile_dim2=0
  g1[5] = stride_d0;                                      // tensor_dim0_stride[31:0]
  g1[6] = 0u;                                             // stride hi, dim1_stride lo
  g1[7] = 0u;
  asm volatile("tensor_load_to_lds %0, %1" :: "s"(g0), "s"(g1) : "memory");
}
__device__ __forceinline__ void wait_tensor0() {
  __builtin_amdgcn_s_wait_tensorcnt((short)0);
}

// ---------------- elementwise conversion kernels ----------------
__global__ __launch_bounds__(256) void k_cvt_f32_f16(const float* __restrict__ src,
                                                     _Float16* __restrict__ dst, int n) {
  int i = (blockIdx.x * 256 + threadIdx.x) * 8;
  if (i + 8 <= n) {
    float4 a = *(const float4*)(src + i);
    float4 b = *(const float4*)(src + i + 4);
    __align__(16) _Float16 o[8] = {(_Float16)a.x, (_Float16)a.y, (_Float16)a.z, (_Float16)a.w,
                                   (_Float16)b.x, (_Float16)b.y, (_Float16)b.z, (_Float16)b.w};
    *(uint4*)(dst + i) = *(const uint4*)o;
  }
}

// W [1024][1024] f32 -> Wt [1024][1024] f16 transposed (Wt[n][k] = W[k][n])
__global__ __launch_bounds__(256) void k_cvt_transpose(const float* __restrict__ W,
                                                       _Float16* __restrict__ Wt) {
  int g = blockIdx.x * 256 + threadIdx.x;
  int k = g >> 10, n = g & 1023;
  Wt[(size_t)n * 1024 + k] = (_Float16)W[g];
}

// ---------------- routing = softmax(x @ Wg + bg) ----------------
__global__ __launch_bounds__(256) void k_routing(const float* __restrict__ X,
                                                 const float* __restrict__ Wg,
                                                 const float* __restrict__ bg,
                                                 float* __restrict__ R) {
  const int lane = threadIdx.x & 31, wave = threadIdx.x >> 5;
  const int row = blockIdx.x * 8 + wave;
  const float* xr = X + (size_t)row * D_;
  float acc[8];
  for (int e = 0; e < 8; ++e) acc[e] = 0.0f;
  for (int k = lane; k < D_; k += 32) {
    float xv = xr[k];
    float4 w0 = *(const float4*)(Wg + k * 8);
    float4 w1 = *(const float4*)(Wg + k * 8 + 4);
    acc[0] += xv * w0.x; acc[1] += xv * w0.y; acc[2] += xv * w0.z; acc[3] += xv * w0.w;
    acc[4] += xv * w1.x; acc[5] += xv * w1.y; acc[6] += xv * w1.z; acc[7] += xv * w1.w;
  }
  for (int o = 16; o > 0; o >>= 1)
    for (int e = 0; e < 8; ++e) acc[e] += __shfl_xor(acc[e], o, 32);
  if (lane == 0) {
    float mx = -3.0e38f;
    for (int e = 0; e < 8; ++e) { acc[e] += bg[e]; mx = fmaxf(mx, acc[e]); }
    float s = 0.0f;
    for (int e = 0; e < 8; ++e) { acc[e] = __expf(acc[e] - mx); s += acc[e]; }
    float inv = 1.0f / s;
    for (int e = 0; e < 8; ++e) R[(size_t)row * 8 + e] = acc[e] * inv;
  }
}

// ---------------- q = expert_emb @ Wq + bq (tiny) ----------------
__global__ __launch_bounds__(256) void k_q(const float* __restrict__ EMB,
                                           const float* __restrict__ Wq,
                                           const float* __restrict__ bq,
                                           float* __restrict__ Q) {
  int g = blockIdx.x * 256 + threadIdx.x; // 8192
  int e = g >> 10, n = g & 1023;
  float acc = bq[n];
  for (int k = 0; k < D_; ++k) acc += EMB[e * D_ + k] * Wq[(size_t)k * D_ + n];
  Q[g] = acc;
}

// ---------------- k/v projections: [32768,1024]x[1024,1024] f16 WMMA ----------------
// A tiles arrive via the Tensor Data Mover (wave 0 issues the D#), B tiles via
// per-lane async global->LDS b128 copies; both double-buffered.
__global__ __launch_bounds__(256) void k_gemm_kv(const _Float16* __restrict__ X,
                                                 const _Float16* __restrict__ Wt0,
                                                 const _Float16* __restrict__ Wt1,
                                                 const float* __restrict__ bias0,
                                                 const float* __restrict__ bias1,
                                                 _Float16* __restrict__ O0,
                                                 _Float16* __restrict__ O1) {
  __shared__ __align__(16) unsigned char sm[32768];   // A[2][128*32] | B[2][128*32]
  _Float16* As = (_Float16*)sm;
  _Float16* Bs = (_Float16*)(sm + 16384);
  const _Float16* Wt  = blockIdx.z ? Wt1 : Wt0;
  const float*   bias = blockIdx.z ? bias1 : bias0;
  _Float16*      O    = blockIdx.z ? O1 : O0;
  const int m0 = blockIdx.y * 128;
  const int n0 = blockIdx.x * 128;
  const int tid = threadIdx.x, lane = tid & 31, wave = tid >> 5;
  const int wm = (wave & 1) * 64;   // 2 waves over M
  const int wn = (wave >> 1) * 32;  // 4 waves over N
  const int r = lane & 15, hi = lane >> 4;

  v8f acc[4][2];
  for (int i = 0; i < 4; ++i) for (int j = 0; j < 2; ++j) acc[i][j] = zero8f();

  auto issueA = [&](int buf, int kc) {  // TDM: tile 32(K) x 128(M) rows of X
    if (wave == 0)
      tdm_load_2d(As + buf * 4096,
                  X + (size_t)m0 * 1024 + (size_t)kc * 32,
                  /*tensor_d0=*/1024, /*tensor_d1=*/32768,
                  /*tile_d0=*/32, /*tile_d1=*/128, /*stride_d0=*/1024);
  };
  auto issueB = [&](int buf, int kc) {  // async lane copies
    for (int c = tid; c < 512; c += 256)
      async_cp16(Bs + buf * 4096 + c * 8,
                 Wt + (size_t)(n0 + (c >> 2)) * 1024 + kc * 32 + (c & 3) * 8);
  };

  issueA(0, 0);
  issueB(0, 0);
  if (wave == 0) wait_tensor0();
  wait_async0();
  __syncthreads();

  for (int kc = 0; kc < 32; ++kc) {
    int buf = kc & 1;
    if (kc + 1 < 32) { issueA(buf ^ 1, kc + 1); issueB(buf ^ 1, kc + 1); }
    const _Float16* A  = As + buf * 4096;
    const _Float16* Bt = Bs + buf * 4096;
    v16h af[4], bf[2];
    for (int mt = 0; mt < 4; ++mt) {     // A 16x32 fragment
      FragU f;
      const _Float16* p = A + (wm + mt * 16 + r) * 32;
      f.q[0] = *(const uint4*)(p + hi * 8);
      f.q[1] = *(const uint4*)(p + 16 + hi * 8);
      af[mt] = f.v;
    }
    for (int nt = 0; nt < 2; ++nt) {     // B 32x16 fragment
      FragU f;
      const _Float16* p = Bt + (wn + nt * 16 + r) * 32;
      f.q[0] = *(const uint4*)(p + hi * 8);
      f.q[1] = *(const uint4*)(p + 16 + hi * 8);
      bf[nt] = f.v;
    }
    for (int mt = 0; mt < 4; ++mt)
      for (int nt = 0; nt < 2; ++nt)
        acc[mt][nt] = wmma_f16(af[mt], bf[nt], acc[mt][nt]);
    if (wave == 0) wait_tensor0();
    wait_async0();
    __syncthreads();
  }

  for (int mt = 0; mt < 4; ++mt)
    for (int nt = 0; nt < 2; ++nt)
      for (int j = 0; j < 8; ++j) {
        int row = m0 + wm + mt * 16 + hi * 8 + j;
        int col = n0 + wn + nt * 16 + r;
        O[(size_t)row * 1024 + col] = (_Float16)(acc[mt][nt][j] + bias[col]);
      }
}

// ---------------- attention per (b,h): scores->softmax->ctx, all in LDS ----------------
__global__ __launch_bounds__(256) void k_attn(const float* __restrict__ Q,       // [8][1024]
                                              const _Float16* __restrict__ K16,  // [B*S][1024]
                                              const _Float16* __restrict__ V16,  // [B*S][1024]
                                              float* __restrict__ CTX) {         // [B][8][1024]
  __shared__ __align__(16) unsigned char sm[65536 + 32768 + 16384 + 64];
  _Float16* P  = (_Float16*)sm;                          // [8][4096] probs (f16)
  _Float16* VT = (_Float16*)(sm + 65536);                // [8 waves][64][32] v^T staging
  float* CR    = (float*)(sm + 65536 + 32768);           // [8 waves][8][64] ctx partials
  float* INV   = (float*)(sm + 65536 + 32768 + 16384);   // [8] 1/sum
  const int h = blockIdx.x, b = blockIdx.y;
  const int hd0 = h * HD_;
  const size_t baseR = (size_t)b * S_;
  const int tid = threadIdx.x, lane = tid & 31, wave = tid >> 5;
  const int m = lane & 15, hi = lane >> 4;

  // q A-fragments (E padded to 16 rows), two K-halves of 32
  v16h qa[2];
  for (int c = 0; c < 2; ++c) {
    FragU f;
    for (int i = 0; i < 8; ++i) {
      f.v[i]     = (m < 8) ? (_Float16)Q[m * D_ + hd0 + c * 32 + hi * 8 + i]      : (_Float16)0.0f;
      f.v[8 + i] = (m < 8) ? (_Float16)Q[m * D_ + hd0 + c * 32 + 16 + hi * 8 + i] : (_Float16)0.0f;
    }
    qa[c] = f.v;
  }

  // phase 1: scores = scale * q @ k^T, wave handles its 512-wide S slice
  const int sbeg = wave * 512;
  for (int sc = 0; sc < 32; ++sc) {
    int s0 = sbeg + sc * 16;
    v8f acc = zero8f();
    for (int c = 0; c < 2; ++c) {
      FragU f;  // B fragment: lane column = s, table-order d per lane
      const _Float16* kp = K16 + (baseR + s0 + m) * D_ + hd0 + c * 32;
      f.q[0] = *(const uint4*)(kp + hi * 8);
      f.q[1] = *(const uint4*)(kp + 16 + hi * 8);
      acc = wmma_f16(qa[c], f.v, acc);
    }
    if (hi == 0)
      for (int j = 0; j < 8; ++j)
        P[j * S_ + s0 + m] = (_Float16)(acc[j] * 0.125f);  // 1/sqrt(64)
  }
  __syncthreads();

  // phase 2: softmax over S, wave w owns expert row e=w
  {
    int e = wave;
    float mx = -3.0e38f;
    for (int s = lane; s < S_; s += 32) mx = fmaxf(mx, (float)P[e * S_ + s]);
    for (int o = 16; o > 0; o >>= 1) mx = fmaxf(mx, __shfl_xor(mx, o, 32));
    float sum = 0.0f;
    for (int s = lane; s < S_; s += 32) {
      float pv = __expf((float)P[e * S_ + s] - mx);
      sum += pv;
      P[e * S_ + s] = (_Float16)pv;
    }
    for (int o = 16; o > 0; o >>= 1) sum += __shfl_xor(sum, o, 32);
    if (lane == 0) INV[e] = 1.0f / sum;
  }
  __syncthreads();

  // phase 3: ctx = attn @ v (per-wave partials over its S slice)
  v8f ct[4];
  for (int i = 0; i < 4; ++i) ct[i] = zero8f();
  _Float16* vt = VT + wave * 64 * 32;
  for (int sc = 0; sc < 16; ++sc) {
    int s0 = sbeg + sc * 32;
    {   // stage v[s0..s0+31][hd0..hd0+63] transposed into LDS
      const _Float16* vp = V16 + (baseR + s0 + lane) * D_ + hd0;
      __align__(16) _Float16 tmp[64];
      for (int q8 = 0; q8 < 8; ++q8)
        *(uint4*)(tmp + q8 * 8) = *(const uint4*)(vp + q8 * 8);
      for (int d = 0; d < 64; ++d) vt[d * 32 + lane] = tmp[d];
    }
    v16h af = zero16h();
    if (m < 8) {
      FragU f;
      const _Float16* pp = P + m * S_ + s0;
      f.q[0] = *(const uint4*)(pp + hi * 8);
      f.q[1] = *(const uint4*)(pp + 16 + hi * 8);
      af = f.v;
    }
    for (int nt = 0; nt < 4; ++nt) {
      FragU f;
      const _Float16* bp = vt + (nt * 16 + m) * 32;
      f.q[0] = *(const uint4*)(bp + hi * 8);
      f.q[1] = *(const uint4*)(bp + 16 + hi * 8);
      ct[nt] = wmma_f16(af, f.v, ct[nt]);
    }
  }
  if (hi == 0)
    for (int nt = 0; nt < 4; ++nt)
      for (int j = 0; j < 8; ++j)
        CR[(wave * 8 + j) * 64 + nt * 16 + m] = ct[nt][j];
  __syncthreads();

  for (int i = tid; i < 512; i += 256) {
    int e = i >> 6, d = i & 63;
    float s = 0.0f;
    for (int w2 = 0; w2 < 8; ++w2) s += CR[(w2 * 8 + e) * 64 + d];
    CTX[((size_t)b * 8 + e) * D_ + hd0 + d] = s * INV[e];
  }
}

// ---------------- attended = ctx @ Wo + bo, f32 WMMA (16x16x4) ----------------
__global__ __launch_bounds__(256) void k_attended(const float* __restrict__ CTX,  // [64][1024]
                                                  const float* __restrict__ Wo,
                                                  const float* __restrict__ bo,
                                                  float* __restrict__ ATT) {      // [64][1024]
  const int n0 = blockIdx.x * 64;
  const int lane = threadIdx.x & 31, wave = threadIdx.x >> 5;
  const int m0 = (wave & 3) * 16;
  const int nb = (wave >> 2) * 32;
  const int m = lane & 15, hi = lane >> 4;
  v8f acc[2] = {zero8f(), zero8f()};
  for (int k = 0; k < D_; k += 4) {
    int kk = k + hi * 2;
    v2f a;
    a.x = CTX[(size_t)(m0 + m) * D_ + kk];
    a.y = CTX[(size_t)(m0 + m) * D_ + kk + 1];
    for (int nt = 0; nt < 2; ++nt) {
      int n = n0 + nb + nt * 16 + m;
      v2f bb;
      bb.x = Wo[(size_t)kk * D_ + n];
      bb.y = Wo[(size_t)(kk + 1) * D_ + n];
      acc[nt] = wmma_f32(a, bb, acc[nt]);
    }
  }
  for (int nt = 0; nt < 2; ++nt)
    for (int j = 0; j < 8; ++j) {
      int row = m0 + hi * 8 + j;
      int col = n0 + nb + nt * 16 + m;
      ATT[(size_t)row * D_ + col] = acc[nt][j] + bo[col];
    }
}

// ---------------- per-expert MLP layer (memory-bound, f32 WMMA) ----------------
__global__ __launch_bounds__(256) void k_mlp(const float* __restrict__ IN,    // [8b x 8e][Kdim]
                                             const float* __restrict__ W,     // [E][Kdim][Ndim]
                                             const float* __restrict__ bias,  // [E][Ndim]
                                             float* __restrict__ OUT,         // [8b x 8e][Ndim]
                                             int Kdim, int Ndim, int relu) {
  const int e = blockIdx.y;
  const int n0 = blockIdx.x * 128;
  const int lane = threadIdx.x & 31, wave = threadIdx.x >> 5;
  const int m = lane & 15, hi = lane >> 4;
  const float* We = W + (size_t)e * Kdim * Ndim;
  const int n = n0 + wave * 16 + m;
  v8f acc = zero8f();
  for (int k = 0; k < Kdim; k += 4) {
    int kk = k + hi * 2;
    v2f a;
    if (m < 8) { a.x = IN[(size_t)(m * 8 + e) * Kdim + kk]; a.y = IN[(size_t)(m * 8 + e) * Kdim + kk + 1]; }
    else       { a.x = 0.0f; a.y = 0.0f; }
    v2f bb;
    bb.x = We[(size_t)kk * Ndim + n];
    bb.y = We[(size_t)(kk + 1) * Ndim + n];
    acc = wmma_f32(a, bb, acc);
  }
  if (hi == 0)
    for (int j = 0; j < 8; ++j) {
      float v = acc[j] + bias[(size_t)e * Ndim + n];
      if (relu) v = fmaxf(v, 0.0f);
      OUT[(size_t)(j * 8 + e) * Ndim + n] = v;
    }
}

// ---------------- out[b,s,:] = sum_e routing[b,s,e] * expert_out[b,e,:] ----------------
__global__ __launch_bounds__(256) void k_combine(const float* __restrict__ R,   // [B][S][8]
                                                 const float* __restrict__ EO,  // [B][8][1024]
                                                 float* __restrict__ OUT) {     // [B][S][1024]
  __shared__ float eo[8 * 1024];
  __shared__ float rr[32 * 8];
  const int b = blockIdx.y;
  const int s0 = blockIdx.x * 32;
  const int tid = threadIdx.x;
  for (int i = tid; i < 2048; i += 256)
    ((float4*)eo)[i] = ((const float4*)(EO + (size_t)b * 8192))[i];
  rr[tid] = R[((size_t)b * S_ + s0 + (tid >> 3)) * 8 + (tid & 7)];
  __syncthreads();
  for (int sl = 0; sl < 32; ++sl) {
    float4 o = {0.0f, 0.0f, 0.0f, 0.0f};
    for (int e = 0; e < 8; ++e) {
      float r = rr[sl * 8 + e];
      float4 v = *(const float4*)(eo + e * 1024 + tid * 4);
      o.x += r * v.x; o.y += r * v.y; o.z += r * v.z; o.w += r * v.w;
    }
    *(float4*)(OUT + ((size_t)b * S_ + s0 + sl) * D_ + tid * 4) = o;
  }
}

extern "C" void kernel_launch(void* const* d_in, const int* in_sizes, int n_in,
                              void* d_out, int out_size, void* d_ws, size_t ws_size,
                              hipStream_t stream) {
  (void)in_sizes; (void)n_in; (void)out_size; (void)ws_size;
  const float* x   = (const float*)d_in[0];
  const float* emb = (const float*)d_in[1];
  const float* Wg  = (const float*)d_in[2];
  const float* bg  = (const float*)d_in[3];
  const float* Wq  = (const float*)d_in[4];
  const float* Wk  = (const float*)d_in[5];
  const float* Wv  = (const float*)d_in[6];
  const float* bq  = (const float*)d_in[7];
  const float* bk  = (const float*)d_in[8];
  const float* bv  = (const float*)d_in[9];
  const float* Wo  = (const float*)d_in[10];
  const float* bo  = (const float*)d_in[11];
  const float* W1  = (const float*)d_in[12];
  const float* b1  = (const float*)d_in[13];
  const float* W2  = (const float*)d_in[14];
  const float* b2  = (const float*)d_in[15];
  float* out = (float*)d_out;

  uint8_t* ws = (uint8_t*)d_ws;
  size_t off = 0;
  auto alloc = [&](size_t bytes) -> void* {
    void* p = ws + off;
    off = (off + bytes + 255) & ~(size_t)255;
    return p;
  };
  const size_t BS = (size_t)B_ * S_;                // 32768
  _Float16* xh  = (_Float16*)alloc(BS * D_ * 2);    // x in f16
  _Float16* wkt = (_Float16*)alloc((size_t)D_ * D_ * 2);
  _Float16* wvt = (_Float16*)alloc((size_t)D_ * D_ * 2);
  _Float16* kh  = (_Float16*)alloc(BS * D_ * 2);
  _Float16* vh  = (_Float16*)alloc(BS * D_ * 2);
  float* qb   = (float*)alloc((size_t)E_ * D_ * 4);
  float* rt   = (float*)alloc(BS * E_ * 4);
  float* ctx  = (float*)alloc((size_t)B_ * E_ * D_ * 4);
  float* att  = (float*)alloc((size_t)B_ * E_ * D_ * 4);
  float* hbuf = (float*)alloc((size_t)B_ * E_ * 2 * D_ * 4);
  float* eo   = (float*)alloc((size_t)B_ * E_ * D_ * 4);

  k_cvt_f32_f16 <<<dim3(16384), 256, 0, stream>>>(x, xh, (int)(BS * D_));
  k_cvt_transpose<<<dim3(4096), 256, 0, stream>>>(Wk, wkt);
  k_cvt_transpose<<<dim3(4096), 256, 0, stream>>>(Wv, wvt);
  k_routing     <<<dim3(4096), 256, 0, stream>>>(x, Wg, bg, rt);
  k_q           <<<dim3(32),   256, 0, stream>>>(emb, Wq, bq, qb);
  k_gemm_kv     <<<dim3(8, 256, 2), 256, 0, stream>>>(xh, wkt, wvt, bk, bv, kh, vh);
  k_attn        <<<dim3(H_, B_), 256, 0, stream>>>(qb, kh, vh, ctx);
  k_attended    <<<dim3(16),  256, 0, stream>>>(ctx, Wo, bo, att);
  k_mlp         <<<dim3(16, 8), 256, 0, stream>>>(att, W1, b1, hbuf, 1024, 2048, 1);
  k_mlp         <<<dim3(8, 8),  256, 0, stream>>>(hbuf, W2, b2, eo, 2048, 1024, 0);
  k_combine     <<<dim3(128, 8), 256, 0, stream>>>(rt, eo, out);
}
